// Block_7696581394616
// MI455X (gfx1250) — matrix operand order repaired and checked by reading, not compiled, not run on hardware
//
#include <hip/hip_runtime.h>
#include <hip/hip_bf16.h>
#include <math.h>

// ---------------------------------------------------------------------------
// ViT block on gfx1250 (MI455X). All large matmuls (QKV, QK^T, PV, proj) run
// on v_wmma_f32_16x16x32_f16 (f16 in / f32 accumulate). Roofline: ~65 GFLOP,
// ~120 MB HBM traffic -> matrix-core bound. GEMM waves are register-blocked
// 32x64 (2 A frags x 4 B frags -> 8 WMMAs per K-step) so the load pipe
// (1.5 b128/WMMA) no longer paces the matrix pipe. Operands stay L2-resident
// (192 MB L2 >> 30 MB working set). MLP hidden=16 is ~0.4 GFLOP -> fp32 VALU.
// ---------------------------------------------------------------------------

#define DIM       768
#define HEADS     12
#define HEAD_DIM  64
#define BATCH     8
#define SEQ       1024
#define TOKENS    (BATCH * SEQ)
#define MLP_HID   16
#define EPS_LN    1e-5f

typedef __attribute__((ext_vector_type(16))) _Float16 v16h;
typedef __attribute__((ext_vector_type(8)))  float    v8f;

union FragH { v16h v; uint32_t u[8]; _Float16 h[16]; };
union FragF { v8f  v; float f[8]; };

// A-matrix 16x32 f16 fragment (row-major source, row stride ld elements).
// Per ISA table: lane half 0 -> K 0..7 (VGPR0-3) and 16..23 (VGPR4-7);
// lane half 1 -> K 8..15 and 24..31. Row M = lane&15.
__device__ __forceinline__ v16h load_fragA_rm(const _Float16* base, int ld) {
  const int lane = threadIdx.x & 31;
  const int half = lane >> 4;
  const int r    = lane & 15;
  const uint32_t* p = (const uint32_t*)(base + (size_t)r * ld);
  FragH f;
#pragma unroll
  for (int j = 0; j < 4; ++j) f.u[j] = p[half * 4 + j];         // K = half*8 + 2j
#pragma unroll
  for (int j = 0; j < 4; ++j) f.u[4 + j] = p[8 + half * 4 + j]; // K = 16 + half*8 + 2j
  return f.v;
}

// B-matrix 32x16 f16 fragment, sourced from a row-major [N][K] array
// (i.e. B stored transposed so K is contiguous). Same K order as A:
// lane half h -> K 8h..8h+7 and 16+8h..16+8h+7; col N = lane&15.
__device__ __forceinline__ v16h load_fragB_rm(const _Float16* base, int ld) {
  const int lane = threadIdx.x & 31;
  const int half = lane >> 4;
  const int r    = lane & 15;
  const uint32_t* p = (const uint32_t*)(base + (size_t)r * ld);
  FragH f;
#pragma unroll
  for (int j = 0; j < 4; ++j) f.u[j] = p[half * 4 + j];         // K = half*8 + 2j
#pragma unroll
  for (int j = 0; j < 4; ++j) f.u[4 + j] = p[8 + half * 4 + j]; // K = 16 + half*8 + 2j
  return f.v;
}

__device__ __forceinline__ v8f wmma16(v16h a, v16h b, v8f c) {
  return __builtin_amdgcn_wmma_f32_16x16x32_f16(false, a, false, b,
                                                (short)0, c, false, false);
}

// ---------------------------------------------------------------------------
// Kernel 1: LayerNorm -> f16
// ---------------------------------------------------------------------------
__global__ void ln_to_f16_kernel(const float* __restrict__ x,
                                 const float* __restrict__ g,
                                 const float* __restrict__ b,
                                 _Float16* __restrict__ out) {
  __shared__ float red[256];
  const int t = blockIdx.x, tid = threadIdx.x;
  const float* row = x + (size_t)t * DIM;
  float v0 = row[tid], v1 = row[tid + 256], v2 = row[tid + 512];
  red[tid] = v0 + v1 + v2; __syncthreads();
  for (int o = 128; o > 0; o >>= 1) { if (tid < o) red[tid] += red[tid + o]; __syncthreads(); }
  float mu = red[0] * (1.0f / DIM); __syncthreads();
  float d0 = v0 - mu, d1 = v1 - mu, d2 = v2 - mu;
  red[tid] = d0 * d0 + d1 * d1 + d2 * d2; __syncthreads();
  for (int o = 128; o > 0; o >>= 1) { if (tid < o) red[tid] += red[tid + o]; __syncthreads(); }
  float rs = rsqrtf(red[0] * (1.0f / DIM) + EPS_LN);
  _Float16* orow = out + (size_t)t * DIM;
  orow[tid]       = (_Float16)(d0 * rs * g[tid]       + b[tid]);
  orow[tid + 256] = (_Float16)(d1 * rs * g[tid + 256] + b[tid + 256]);
  orow[tid + 512] = (_Float16)(d2 * rs * g[tid + 512] + b[tid + 512]);
}

// ---------------------------------------------------------------------------
// Kernel 2: fp32 [R][C] -> f16 [C][R] (weights transposed for B fragments)
// ---------------------------------------------------------------------------
__global__ void transpose_to_f16_kernel(const float* __restrict__ in,
                                        _Float16* __restrict__ out,
                                        int R, int C) {
  size_t n = (size_t)R * C;
  for (size_t idx = (size_t)blockIdx.x * blockDim.x + threadIdx.x; idx < n;
       idx += (size_t)gridDim.x * blockDim.x) {
    size_t c = idx / R, r = idx % R;
    out[idx] = (_Float16)in[r * (size_t)C + c];
  }
}

// ---------------------------------------------------------------------------
// Kernel 3: QKV GEMM [8192x2304] = Hf @ WqkvT^T, K=768.
// One wave per 32x64 output tile: 2 A frags x 4 B frags -> 8 WMMAs/K-step.
// Column split is the reference's faithful quirk: c = which*768 + d*12 + h.
// Q,K stored [bh][p][64] row-major; V stored transposed [bh][64][p].
// ---------------------------------------------------------------------------
__global__ void qkv_gemm_kernel(const _Float16* __restrict__ Hf,
                                const _Float16* __restrict__ WT,  // [2304][768]
                                const float* __restrict__ bias,
                                _Float16* __restrict__ Q,
                                _Float16* __restrict__ K,
                                _Float16* __restrict__ Vt) {
  const int wave = threadIdx.x >> 5;
  const int lane = threadIdx.x & 31;
  const int half = lane >> 4;
  const int l16  = lane & 15;
  const int n0 = (blockIdx.x * 4 + wave) * 64;   // 64 cols per wave
  const int m0 = blockIdx.y * 32;                // 32 rows per wave
  const v8f vzero = {};
  v8f acc[2][4];
#pragma unroll
  for (int mi = 0; mi < 2; ++mi)
#pragma unroll
    for (int ni = 0; ni < 4; ++ni) acc[mi][ni] = vzero;
  const _Float16* aBase = Hf + (size_t)m0 * DIM;
  const _Float16* bBase = WT + (size_t)n0 * DIM;
  for (int kk = 0; kk < DIM; kk += 32) {
    v16h a0 = load_fragA_rm(aBase + kk, DIM);
    v16h a1 = load_fragA_rm(aBase + 16 * DIM + kk, DIM);
    v16h b0 = load_fragB_rm(bBase + kk, DIM);
    v16h b1 = load_fragB_rm(bBase + 16 * DIM + kk, DIM);
    v16h b2 = load_fragB_rm(bBase + 32 * DIM + kk, DIM);
    v16h b3 = load_fragB_rm(bBase + 48 * DIM + kk, DIM);
    acc[0][0] = wmma16(a0, b0, acc[0][0]);
    acc[0][1] = wmma16(a0, b1, acc[0][1]);
    acc[0][2] = wmma16(a0, b2, acc[0][2]);
    acc[0][3] = wmma16(a0, b3, acc[0][3]);
    acc[1][0] = wmma16(a1, b0, acc[1][0]);
    acc[1][1] = wmma16(a1, b1, acc[1][1]);
    acc[1][2] = wmma16(a1, b2, acc[1][2]);
    acc[1][3] = wmma16(a1, b3, acc[1][3]);
  }
#pragma unroll
  for (int ni = 0; ni < 4; ++ni) {
    const int col   = n0 + ni * 16 + l16;
    const int which = col / DIM;
    const int r     = col % DIM;
    const int d     = r / HEADS;   // head_dim index (source split order)
    const int hh    = r % HEADS;   // head index
    const float bv  = bias[col];
#pragma unroll
    for (int mi = 0; mi < 2; ++mi) {
      FragF c; c.v = acc[mi][ni];
#pragma unroll
      for (int i = 0; i < 8; ++i) {
        int t = m0 + mi * 16 + i + 8 * half;
        int bimg = t >> 10, p = t & 1023;
        size_t bh = (size_t)bimg * HEADS + hh;
        _Float16 val = (_Float16)(c.f[i] + bv);
        if (which == 0)      Q[(bh * SEQ + p) * HEAD_DIM + d]  = val;
        else if (which == 1) K[(bh * SEQ + p) * HEAD_DIM + d]  = val;
        else                 Vt[(bh * HEAD_DIM + d) * SEQ + p] = val;
      }
    }
  }
}

// ---------------------------------------------------------------------------
// Kernel 4: flash attention, one wave per (b,h, 16-query tile).
// QK^T and PV on WMMA; online softmax with shfl_xor reductions across the
// 16-lane half that owns each C-layout row; P goes C-layout -> A-layout
// through a wave-private LDS tile. Next K/V tile prefetched (global_prefetch).
// ---------------------------------------------------------------------------
__global__ void attn_kernel(const _Float16* __restrict__ Q,
                            const _Float16* __restrict__ K,
                            const _Float16* __restrict__ Vt,
                            _Float16* __restrict__ O) {  // [T][768], col = h*64+d
  __shared__ _Float16 pTile[16 * 32];
  const int lane = threadIdx.x & 31;
  const int half = lane >> 4;
  const int l16  = lane & 15;
  const int bh = blockIdx.x >> 6;
  const int m0 = (blockIdx.x & 63) << 4;
  const _Float16* Qb = Q  + ((size_t)bh * SEQ + m0) * HEAD_DIM;
  const _Float16* Kb = K  + (size_t)bh * SEQ * HEAD_DIM;
  const _Float16* Vb = Vt + (size_t)bh * HEAD_DIM * SEQ;

  v16h q0 = load_fragA_rm(Qb, HEAD_DIM);
  v16h q1 = load_fragA_rm(Qb + 32, HEAD_DIM);

  v8f o0 = {}, o1 = {}, o2 = {}, o3 = {};
  float mrun[8], lrun[8];
#pragma unroll
  for (int i = 0; i < 8; ++i) { mrun[i] = -3.0e38f; lrun[i] = 0.0f; }
  const float scale = 0.125f;  // 64^-0.5

  for (int j0 = 0; j0 < SEQ; j0 += 32) {
    // Pull the next 32-key K tile and next V columns toward the WGP while the
    // current tile's WMMA + softmax VALU run (global_prefetch_b8 path).
    if (j0 + 32 < SEQ) {
      __builtin_prefetch(Kb + (size_t)(j0 + 32 + lane) * HEAD_DIM, 0, 3);
      __builtin_prefetch(Vb + (size_t)lane * SEQ + j0 + 32, 0, 3);
      __builtin_prefetch(Vb + (size_t)(lane + 32) * SEQ + j0 + 32, 0, 3);
    }
    v8f sa = {}, sb = {};
    sa = wmma16(q0, load_fragB_rm(Kb + (size_t)j0 * HEAD_DIM,      HEAD_DIM), sa);
    sa = wmma16(q1, load_fragB_rm(Kb + (size_t)j0 * HEAD_DIM + 32, HEAD_DIM), sa);
    sb = wmma16(q0, load_fragB_rm(Kb + (size_t)(j0 + 16) * HEAD_DIM,      HEAD_DIM), sb);
    sb = wmma16(q1, load_fragB_rm(Kb + (size_t)(j0 + 16) * HEAD_DIM + 32, HEAD_DIM), sb);
    FragF A; A.v = sa; FragF B; B.v = sb;
    float fsc[8];
#pragma unroll
    for (int i = 0; i < 8; ++i) {
      float x0 = A.f[i] * scale, x1 = B.f[i] * scale;
      float mx = fmaxf(x0, x1);
      mx = fmaxf(mx, __shfl_xor(mx, 1));
      mx = fmaxf(mx, __shfl_xor(mx, 2));
      mx = fmaxf(mx, __shfl_xor(mx, 4));
      mx = fmaxf(mx, __shfl_xor(mx, 8));
      float mnew = fmaxf(mrun[i], mx);
      float f  = __expf(mrun[i] - mnew);
      float p0 = __expf(x0 - mnew);
      float p1 = __expf(x1 - mnew);
      float rs = p0 + p1;
      rs += __shfl_xor(rs, 1); rs += __shfl_xor(rs, 2);
      rs += __shfl_xor(rs, 4); rs += __shfl_xor(rs, 8);
      lrun[i] = lrun[i] * f + rs;
      mrun[i] = mnew;
      fsc[i] = f;
      pTile[(i + 8 * half) * 32 + l16]      = (_Float16)p0;
      pTile[(i + 8 * half) * 32 + 16 + l16] = (_Float16)p1;
    }
    FragF t0, t1, t2, t3; t0.v = o0; t1.v = o1; t2.v = o2; t3.v = o3;
#pragma unroll
    for (int i = 0; i < 8; ++i) {
      t0.f[i] *= fsc[i]; t1.f[i] *= fsc[i]; t2.f[i] *= fsc[i]; t3.f[i] *= fsc[i];
    }
    __syncthreads();                       // single-wave block: cheap LDS fence
    v16h pf = load_fragA_rm(pTile, 32);
    __syncthreads();
    o0 = wmma16(pf, load_fragB_rm(Vb + 0 * 16 * SEQ + j0, SEQ), t0.v);
    o1 = wmma16(pf, load_fragB_rm(Vb + 1 * 16 * SEQ + j0, SEQ), t1.v);
    o2 = wmma16(pf, load_fragB_rm(Vb + 2 * 16 * SEQ + j0, SEQ), t2.v);
    o3 = wmma16(pf, load_fragB_rm(Vb + 3 * 16 * SEQ + j0, SEQ), t3.v);
  }

  const int b = bh / HEADS, h = bh % HEADS;
  FragF t0, t1, t2, t3; t0.v = o0; t1.v = o1; t2.v = o2; t3.v = o3;
#pragma unroll
  for (int i = 0; i < 8; ++i) {
    int p = m0 + i + 8 * half;
    size_t t = (size_t)b * SEQ + p;
    float inv = 1.0f / lrun[i];
    O[t * DIM + h * HEAD_DIM + 0 * 16 + l16] = (_Float16)(t0.f[i] * inv);
    O[t * DIM + h * HEAD_DIM + 1 * 16 + l16] = (_Float16)(t1.f[i] * inv);
    O[t * DIM + h * HEAD_DIM + 2 * 16 + l16] = (_Float16)(t2.f[i] * inv);
    O[t * DIM + h * HEAD_DIM + 3 * 16 + l16] = (_Float16)(t3.f[i] * inv);
  }
}

// ---------------------------------------------------------------------------
// Kernel 5: proj GEMM + bias + residual -> d_out (fp32).
// Same 32x64-per-wave register blocking as the QKV GEMM.
// ---------------------------------------------------------------------------
__global__ void proj_gemm_kernel(const _Float16* __restrict__ O,
                                 const _Float16* __restrict__ WT,  // [768][768]
                                 const float* __restrict__ bias,
                                 const float* __restrict__ x,
                                 float* __restrict__ out) {
  const int wave = threadIdx.x >> 5;
  const int lane = threadIdx.x & 31;
  const int half = lane >> 4;
  const int l16  = lane & 15;
  const int n0 = (blockIdx.x * 4 + wave) * 64;
  const int m0 = blockIdx.y * 32;
  const v8f vzero = {};
  v8f acc[2][4];
#pragma unroll
  for (int mi = 0; mi < 2; ++mi)
#pragma unroll
    for (int ni = 0; ni < 4; ++ni) acc[mi][ni] = vzero;
  const _Float16* aBase = O  + (size_t)m0 * DIM;
  const _Float16* bBase = WT + (size_t)n0 * DIM;
  for (int kk = 0; kk < DIM; kk += 32) {
    v16h a0 = load_fragA_rm(aBase + kk, DIM);
    v16h a1 = load_fragA_rm(aBase + 16 * DIM + kk, DIM);
    v16h b0 = load_fragB_rm(bBase + kk, DIM);
    v16h b1 = load_fragB_rm(bBase + 16 * DIM + kk, DIM);
    v16h b2 = load_fragB_rm(bBase + 32 * DIM + kk, DIM);
    v16h b3 = load_fragB_rm(bBase + 48 * DIM + kk, DIM);
    acc[0][0] = wmma16(a0, b0, acc[0][0]);
    acc[0][1] = wmma16(a0, b1, acc[0][1]);
    acc[0][2] = wmma16(a0, b2, acc[0][2]);
    acc[0][3] = wmma16(a0, b3, acc[0][3]);
    acc[1][0] = wmma16(a1, b0, acc[1][0]);
    acc[1][1] = wmma16(a1, b1, acc[1][1]);
    acc[1][2] = wmma16(a1, b2, acc[1][2]);
    acc[1][3] = wmma16(a1, b3, acc[1][3]);
  }
#pragma unroll
  for (int ni = 0; ni < 4; ++ni) {
    const int col  = n0 + ni * 16 + l16;
    const float bv = bias[col];
#pragma unroll
    for (int mi = 0; mi < 2; ++mi) {
      FragF c; c.v = acc[mi][ni];
#pragma unroll
      for (int i = 0; i < 8; ++i) {
        size_t t = (size_t)m0 + mi * 16 + i + 8 * half;
        out[t * DIM + col] = x[t * DIM + col] + c.f[i] + bv;
      }
    }
  }
}

// ---------------------------------------------------------------------------
// Kernel 6: LN2 + MLP(16, exact GELU) + residual, in-place on d_out.
// ---------------------------------------------------------------------------
__global__ void ln_mlp_kernel(const float* __restrict__ g2,
                              const float* __restrict__ b2,
                              const float* __restrict__ w_fc1,
                              const float* __restrict__ b_fc1,
                              const float* __restrict__ w_fc2,
                              const float* __restrict__ b_fc2,
                              float* __restrict__ out) {
  __shared__ float red[256];
  __shared__ float lnrow[DIM];
  __shared__ float hact[MLP_HID];
  const int t = blockIdx.x, tid = threadIdx.x;
  float* row = out + (size_t)t * DIM;
  float v0 = row[tid], v1 = row[tid + 256], v2 = row[tid + 512];
  red[tid] = v0 + v1 + v2; __syncthreads();
  for (int o = 128; o > 0; o >>= 1) { if (tid < o) red[tid] += red[tid + o]; __syncthreads(); }
  float mu = red[0] * (1.0f / DIM); __syncthreads();
  float d0 = v0 - mu, d1 = v1 - mu, d2 = v2 - mu;
  red[tid] = d0 * d0 + d1 * d1 + d2 * d2; __syncthreads();
  for (int o = 128; o > 0; o >>= 1) { if (tid < o) red[tid] += red[tid + o]; __syncthreads(); }
  float rs = rsqrtf(red[0] * (1.0f / DIM) + EPS_LN);
  lnrow[tid]       = d0 * rs * g2[tid]       + b2[tid];
  lnrow[tid + 256] = d1 * rs * g2[tid + 256] + b2[tid + 256];
  lnrow[tid + 512] = d2 * rs * g2[tid + 512] + b2[tid + 512];
  __syncthreads();
  if (tid < MLP_HID) {
    float acc = b_fc1[tid];
    for (int k = 0; k < DIM; ++k) acc += lnrow[k] * w_fc1[k * MLP_HID + tid];
    hact[tid] = 0.5f * acc * (1.0f + erff(acc * 0.70710678118654752f)); // exact GELU
  }
  __syncthreads();
#pragma unroll
  for (int j = 0; j < 3; ++j) {
    int c = tid + j * 256;
    float s = b_fc2[c];
#pragma unroll
    for (int jj = 0; jj < MLP_HID; ++jj) s += hact[jj] * w_fc2[jj * DIM + c];
    row[c] += s;
  }
}

// ---------------------------------------------------------------------------
extern "C" void kernel_launch(void* const* d_in, const int* in_sizes, int n_in,
                              void* d_out, int out_size, void* d_ws, size_t ws_size,
                              hipStream_t stream) {
  (void)in_sizes; (void)n_in; (void)out_size; (void)ws_size;
  const float* x      = (const float*)d_in[0];
  const float* g1     = (const float*)d_in[1];
  const float* b1     = (const float*)d_in[2];
  const float* w_qkv  = (const float*)d_in[3];
  const float* b_qkv  = (const float*)d_in[4];
  const float* w_proj = (const float*)d_in[5];
  const float* b_proj = (const float*)d_in[6];
  const float* g2     = (const float*)d_in[7];
  const float* b2     = (const float*)d_in[8];
  const float* w_fc1  = (const float*)d_in[9];
  const float* b_fc1  = (const float*)d_in[10];
  const float* w_fc2  = (const float*)d_in[11];
  const float* b_fc2  = (const float*)d_in[12];
  float* out = (float*)d_out;

  // Workspace layout (~55 MB). The LN1 buffer is re-used as the attention
  // output buffer once the QKV GEMM has consumed it (stream-ordered).
  char* ws = (char*)d_ws;
  size_t off = 0;
  auto take = [&](size_t bytes) {
    void* p = ws + off;
    off += (bytes + 255) & ~(size_t)255;
    return p;
  };
  _Float16* Hf     = (_Float16*)take((size_t)TOKENS * DIM * 2);     // LN1 out / attn O
  _Float16* WqkvT  = (_Float16*)take((size_t)3 * DIM * DIM * 2);    // [2304][768]
  _Float16* WprojT = (_Float16*)take((size_t)DIM * DIM * 2);        // [768][768]
  _Float16* Qh     = (_Float16*)take((size_t)BATCH * HEADS * SEQ * HEAD_DIM * 2);
  _Float16* Kh     = (_Float16*)take((size_t)BATCH * HEADS * SEQ * HEAD_DIM * 2);
  _Float16* Vth    = (_Float16*)take((size_t)BATCH * HEADS * HEAD_DIM * SEQ * 2);
  _Float16* Of     = Hf;

  ln_to_f16_kernel<<<TOKENS, 256, 0, stream>>>(x, g1, b1, Hf);
  transpose_to_f16_kernel<<<2048, 256, 0, stream>>>(w_qkv, WqkvT, DIM, 3 * DIM);
  transpose_to_f16_kernel<<<1024, 256, 0, stream>>>(w_proj, WprojT, DIM, DIM);
  qkv_gemm_kernel<<<dim3(9, 256), 128, 0, stream>>>(Hf, WqkvT, b_qkv, Qh, Kh, Vth);
  attn_kernel<<<BATCH * HEADS * (SEQ / 16), 32, 0, stream>>>(Qh, Kh, Vth, Of);
  proj_gemm_kernel<<<dim3(3, 256), 128, 0, stream>>>(Of, WprojT, b_proj, x, out);
  ln_mlp_kernel<<<TOKENS, 256, 0, stream>>>(g2, b2, w_fc1, b_fc1, w_fc2, b_fc2, out);
}
